// MLP_20744692040155
// MI455X (gfx1250) — matrix operand order repaired and checked by reading, not compiled, not run on hardware
//
#include <hip/hip_runtime.h>
#include <stdint.h>

// ---------------------------------------------------------------------------
// Fused MLP for MI455X (gfx1250, wave32, WMMA), transposed formulation:
//   Phase A:  h^T tiles = Wcat(A) x xvec(B)   -> h stored to LDS in B-operand
//             layout with ONE ds_store_b128 per tile per lane.
//   Phase B:  h2^T = W2(A) x h(B), bias+relu, fused W3 reduction per lane,
//             single cross-half shuffle at the end.
// W2 is staged through a double-buffered LDS slab with async-DMA overlap.
// LDS = 4 waves * 2 subtiles * 32KB (h) + 2 * 32KB (W2 slabs) = 320 KB.
// ---------------------------------------------------------------------------

typedef __attribute__((ext_vector_type(16))) __bf16       v16bf;
typedef __attribute__((ext_vector_type(2)))  __bf16       v2bf;
typedef __attribute__((ext_vector_type(8)))  float        v8f;
typedef __attribute__((ext_vector_type(4)))  unsigned int u32x4;
typedef __attribute__((ext_vector_type(4)))  int          i32x4;
typedef __attribute__((ext_vector_type(4)))  float        f32x4;

#define Hdim 1024
#define NTILES 64           // Hdim / 16 tiles
#define KTILES 32           // Hdim / 32 k tiles
#define SUBTILES 2          // row-tiles per wave
#define ROWS_PER_BLOCK 128  // 4 waves * 2 subtiles * 16 rows
#define THREADS 128
#define SLAB_BASE 262144

#if defined(__AMDGCN__) && __has_builtin(__builtin_amdgcn_global_load_async_to_lds_b128)
#define HAVE_ASYNC 1
#else
#define HAVE_ASYNC 0
#endif

#if HAVE_ASYNC
#if __has_builtin(__builtin_amdgcn_s_wait_asynccnt)
#define WAIT_ASYNC(n) __builtin_amdgcn_s_wait_asynccnt(n)
#else
#define WAIT_ASYNC(n) asm volatile("s_wait_asynccnt %0" ::"i"(n) : "memory")
#endif
#else
#define WAIT_ASYNC(n) ((void)0)
#endif

union BF16Tile {
  v16bf          v;
  unsigned short s[16];
  u32x4          q[2];
};

__device__ __forceinline__ unsigned short f2bf(float f) {
  unsigned u = __float_as_uint(f);
  u += 0x7fffu + ((u >> 16) & 1u);   // round-to-nearest-even
  return (unsigned short)(u >> 16);
}

__device__ __forceinline__ unsigned pk2bf(float a, float b) {
#if defined(__AMDGCN__) && __has_builtin(__builtin_amdgcn_cvt_pk_bf16_f32)
  union { v2bf v; unsigned u; } c;
  c.v = __builtin_amdgcn_cvt_pk_bf16_f32(a, b);   // v_cvt_pk_bf16_f32
  return c.u;
#else
  return (unsigned)f2bf(a) | ((unsigned)f2bf(b) << 16);
#endif
}

// --- prep kernel 1: W2 f32 -> bf16, row-major [n][k] ------------------------
__global__ void prep_w2(const float* __restrict__ W2, unsigned short* __restrict__ w2bf) {
  int i = (blockIdx.x * blockDim.x + threadIdx.x) * 4;
  f32x4 f = *(const f32x4*)(W2 + i);
  unsigned* p = (unsigned*)(w2bf + i);
  p[0] = pk2bf(f.x, f.y);
  p[1] = pk2bf(f.z, f.w);
}

// --- prep kernel 2: Wcat_t[j][k] (bf16), k = [Wmeta(10); bmeta(10); W1; b1; 0*10]
__global__ void prep_wcat(const float* __restrict__ Wmeta, const float* __restrict__ bmeta,
                          const float* __restrict__ W1, const float* __restrict__ b1,
                          unsigned short* __restrict__ wcat) {
  int j = blockIdx.x * blockDim.x + threadIdx.x;
  if (j >= Hdim) return;
  unsigned short row[32];
#pragma unroll
  for (int e = 0; e < 10; ++e) {
    row[e]      = f2bf(Wmeta[e * Hdim + j]);
    row[10 + e] = f2bf(bmeta[e * Hdim + j]);
  }
  row[20] = f2bf(W1[j]);
  row[21] = f2bf(b1[j]);
#pragma unroll
  for (int k = 22; k < 32; ++k) row[k] = 0;
#pragma unroll
  for (int k = 0; k < 32; ++k) wcat[j * 32 + k] = row[k];
}

// --- fused main kernel ------------------------------------------------------
__launch_bounds__(THREADS)
__global__ void fused_mlp(const float* __restrict__ x, const float* __restrict__ b2,
                          const float* __restrict__ W3, const float* __restrict__ b3,
                          const unsigned short* __restrict__ w2bf,
                          const unsigned short* __restrict__ wcat,
                          float* __restrict__ out) {
  __shared__ __align__(16) unsigned char smem[327680];

  const int tid  = threadIdx.x;
  const int lane = tid & 31;
  const int wave = tid >> 5;
  const int half = lane >> 4;
  const int l16  = lane & 15;

  unsigned char* hbase = smem + wave * (SUBTILES * 32768); // [t][kt][lane][32B] B-layout

  const int row0 = blockIdx.x * ROWS_PER_BLOCK + wave * (SUBTILES * 16);

  // ---- W2 slab staging (A-operand layout, 16B units, async when available) --
  // Slab `buf` lives at smem + SLAB_BASE + buf*32768, layout [kt][lane][32B].
  auto stage = [&](int buf, int nt2x) {
    unsigned char* dst = smem + SLAB_BASE + buf * 32768;
#pragma unroll
    for (int r = 0; r < 16; ++r) {
      int u   = tid + THREADS * r;   // 0..2047
      int ukt = u >> 6;
      int ul  = (u >> 1) & 31;
      int uh2 = u & 1;
      const unsigned short* g = w2bf + (size_t)(nt2x * 16 + (ul & 15)) * Hdim
                              + ukt * 32 + (ul >> 4) * 16 + uh2 * 8;
      unsigned char* l = dst + ukt * 1024 + ul * 32 + uh2 * 16;
#if HAVE_ASYNC
      __builtin_amdgcn_global_load_async_to_lds_b128(
          (__attribute__((address_space(1))) i32x4*)(i32x4*)(g),
          (__attribute__((address_space(3))) i32x4*)(i32x4*)(l), 0, 0);
#else
      *(u32x4*)l = *(const u32x4*)g;
#endif
    }
  };

  // Kick off slab 0 DMA; it overlaps all of Phase A.
  stage(0, 0);

  // ---------------- Phase A: layer 1 (transposed) --------------------------
  // B operand per subtile: lane -> column m = l16, K slice = half*16..+15 of
  // [phi*oh(10), oh(10), delta, 1, 0*10] for row (row0 + t*16 + l16).
  BF16Tile bx[SUBTILES];
#pragma unroll
  for (int t = 0; t < SUBTILES; ++t) {
    const float* xr = x + (size_t)(row0 + t * 16 + l16) * 12;
    float xv[12];
#pragma unroll
    for (int i = 0; i < 12; ++i) xv[i] = xr[i];
    const float delta = xv[10], phi = xv[11];
    auto kv = [&](int k) -> float {
      if (k < 10)  return phi * xv[k];
      if (k < 20)  return xv[k - 10];
      if (k == 20) return delta;
      if (k == 21) return 1.0f;
      return 0.0f;
    };
#pragma unroll
    for (int i = 0; i < 16; ++i) bx[t].s[i] = f2bf(kv(half * 16 + i));
  }

  for (int nt = 0; nt < NTILES; ++nt) {
    BF16Tile aop;  // A operand from Wcat: lane -> row j = nt*16 + l16, chunked K
    {
      const unsigned char* wb = (const unsigned char*)wcat + (size_t)(nt * 16 + l16) * 64;
      aop.q[0] = *(const u32x4*)(wb + half * 32);
      aop.q[1] = *(const u32x4*)(wb + half * 32 + 16);
    }
#pragma unroll
    for (int t = 0; t < SUBTILES; ++t) {
      v8f acc = {};
      acc = __builtin_amdgcn_wmma_f32_16x16x32_bf16(false, aop.v, false, bx[t].v,
                                                    (short)0, acc, false, false);
      // acc[v] = h_pre[j = nt*16 + v + 8*half][m = l16]; relu+pack, one b128 store
      u32x4 pk;
      pk.x = pk2bf(fmaxf(acc[0], 0.0f), fmaxf(acc[1], 0.0f));
      pk.y = pk2bf(fmaxf(acc[2], 0.0f), fmaxf(acc[3], 0.0f));
      pk.z = pk2bf(fmaxf(acc[4], 0.0f), fmaxf(acc[5], 0.0f));
      pk.w = pk2bf(fmaxf(acc[6], 0.0f), fmaxf(acc[7], 0.0f));
      int off = t * 32768 + (nt >> 1) * 1024 + (l16 + 16 * (nt & 1)) * 32 + half * 16;
      *(u32x4*)(hbase + off) = pk;
    }
  }

  // ---------------- Phase B: h2^T = W2 x h^T, fused layer 3 ----------------
  float o0[SUBTILES], o1[SUBTILES];
#pragma unroll
  for (int t = 0; t < SUBTILES; ++t) { o0[t] = 0.0f; o1[t] = 0.0f; }

  for (int nt2 = 0; nt2 < NTILES; ++nt2) {
    const int cur = nt2 & 1;
    __syncthreads();                 // everyone finished reading slab[cur^1]
    if (nt2 + 1 < NTILES) {
      stage(cur ^ 1, nt2 + 1);       // next slab DMA overlaps this compute
      WAIT_ASYNC(16);                // slab[cur]'s 16 asyncs (oldest) are done
    } else {
      WAIT_ASYNC(0);
    }
    __syncthreads();                 // slab[cur] visible to all waves
    if (nt2 + 2 < NTILES)
      __builtin_prefetch(w2bf + (size_t)(nt2 + 2) * 16 * Hdim + tid * 128, 0, 1);

    unsigned char* slabcur = smem + SLAB_BASE + cur * 32768;
    v8f acc[SUBTILES] = {};
#pragma unroll 2
    for (int kt = 0; kt < KTILES; ++kt) {
      BF16Tile a2;  // W2 tile (A operand), shared by both subtiles
      const u32x4* ap = (const u32x4*)(slabcur + kt * 1024 + lane * 32);
      a2.q[0] = ap[0];
      a2.q[1] = ap[1];
#pragma unroll
      for (int t = 0; t < SUBTILES; ++t) {
        BF16Tile hb;  // h tile (B operand)
        const u32x4* hp = (const u32x4*)(hbase + t * 32768 + kt * 1024 + lane * 32);
        hb.q[0] = hp[0];
        hb.q[1] = hp[1];
        acc[t] = __builtin_amdgcn_wmma_f32_16x16x32_bf16(false, a2.v, false, hb.v,
                                                         (short)0, acc[t], false, false);
      }
    }

    // acc[t][v]: n = nt2*16 + v + 8*half, m = l16. Bias+relu, dot with W3 rows.
    const float* bpv = b2 + nt2 * 16 + 8 * half;
    f32x4 b2lo = *(const f32x4*)bpv,          b2hi = *(const f32x4*)(bpv + 4);
    const float* w3p = W3 + nt2 * 16 + 8 * half;
    f32x4 w3a0 = *(const f32x4*)w3p,          w3a1 = *(const f32x4*)(w3p + 4);
    f32x4 w3b0 = *(const f32x4*)(w3p + Hdim), w3b1 = *(const f32x4*)(w3p + Hdim + 4);
#pragma unroll
    for (int t = 0; t < SUBTILES; ++t) {
#pragma unroll
      for (int v = 0; v < 8; ++v) {
        float bias = (v < 4) ? b2lo[v & 3] : b2hi[v & 3];
        float wa   = (v < 4) ? w3a0[v & 3] : w3a1[v & 3];
        float wb   = (v < 4) ? w3b0[v & 3] : w3b1[v & 3];
        float hv   = fmaxf(acc[t][v] + bias, 0.0f);
        o0[t] += hv * wa;
        o1[t] += hv * wb;
      }
    }
  }

  // combine the two n-halves (lane l and l+16 hold the same output row m=l16)
#pragma unroll
  for (int t = 0; t < SUBTILES; ++t) {
    o0[t] += __shfl_xor(o0[t], 16, 32);
    o1[t] += __shfl_xor(o1[t], 16, 32);
  }
  if (half == 0) {
    float bb0 = b3[0], bb1 = b3[1];
#pragma unroll
    for (int t = 0; t < SUBTILES; ++t) {
      int row = row0 + t * 16 + l16;
      out[row * 2 + 0] = o0[t] + bb0;
      out[row * 2 + 1] = o1[t] + bb1;
    }
  }
}

// ---------------------------------------------------------------------------
extern "C" void kernel_launch(void* const* d_in, const int* in_sizes, int n_in,
                              void* d_out, int out_size, void* d_ws, size_t ws_size,
                              hipStream_t stream) {
  const float* x     = (const float*)d_in[0];
  const float* W1    = (const float*)d_in[1];
  const float* b1    = (const float*)d_in[2];
  const float* Wmeta = (const float*)d_in[3];
  const float* bmeta = (const float*)d_in[4];
  const float* W2    = (const float*)d_in[5];
  const float* b2    = (const float*)d_in[6];
  const float* W3    = (const float*)d_in[7];
  const float* b3    = (const float*)d_in[8];
  float* out = (float*)d_out;

  unsigned short* w2bf = (unsigned short*)d_ws;                                    // 2 MB
  unsigned short* wcat = (unsigned short*)((char*)d_ws + (size_t)Hdim * Hdim * 2); // 64 KB

  const int Bn = in_sizes[0] / 12;

  prep_w2<<<(Hdim * Hdim) / (256 * 4), 256, 0, stream>>>(W2, w2bf);
  prep_wcat<<<Hdim / 256, 256, 0, stream>>>(Wmeta, bmeta, W1, b1, wcat);
  fused_mlp<<<Bn / ROWS_PER_BLOCK, THREADS, 0, stream>>>(x, b2, W3, b3, w2bf, wcat, out);
}
